// MoELayer_64819646432102
// MI455X (gfx1250) — compile-verified, matrix operand order checked by reading, not run on hardware
//
#include <hip/hip_runtime.h>
#include <hip/hip_bf16.h>
#include <cstddef>

typedef __attribute__((ext_vector_type(16))) _Float16 v16h;
typedef __attribute__((ext_vector_type(8)))  _Float16 v8h;
typedef __attribute__((ext_vector_type(4)))  float    v4f;
typedef __attribute__((ext_vector_type(8)))  float    v8f;

#define MOE_E 8
#define MOE_H 2048
#define MOE_F 4096
#define MOE_M 32   // tokens per FFN block (two 16-row WMMA M-tiles)

// ---------------------------------------------------------------------------
// Kernel 0: one-shot fp32 -> f16 conversion of activations (amortized over
// the 64 reuses of each token row in the FFN kernel).
// ---------------------------------------------------------------------------
__global__ __launch_bounds__(256) void cvt_x_kernel(
    const float* __restrict__ x, _Float16* __restrict__ xh, size_t n)
{
    const size_t i = ((size_t)blockIdx.x * blockDim.x + threadIdx.x) * 8;
    if (i >= n) return;
    const v4f a = *(const v4f*)(x + i);
    const v4f b = *(const v4f*)(x + i + 4);
    union { v8h v; _Float16 s[8]; } u;
#pragma unroll
    for (int j = 0; j < 4; ++j) {
        u.s[j]     = (_Float16)a[j];
        u.s[4 + j] = (_Float16)b[j];
    }
    *(v8h*)(xh + i) = u.v;
}

// ---------------------------------------------------------------------------
// Kernel 1: router. One wave32 per token. 8 logits, top-2, renormalized
// weights; writes router_weights to d_out tail and appends (token, weight)
// to the chosen experts' lists in workspace.
// ---------------------------------------------------------------------------
__global__ __launch_bounds__(256) void moe_gate_kernel(
    const float* __restrict__ x, const float* __restrict__ gw,
    float* __restrict__ rw_out, int* __restrict__ counts,
    int* __restrict__ list, float* __restrict__ lwgt, int T)
{
    const int lane = threadIdx.x & 31;
    const int tok  = blockIdx.x * (blockDim.x >> 5) + (threadIdx.x >> 5);
    if (tok >= T) return;

    float acc[MOE_E];
#pragma unroll
    for (int e = 0; e < MOE_E; ++e) acc[e] = 0.0f;

    const float* xr = x + (size_t)tok * MOE_H;
    for (int k = lane; k < MOE_H; k += 32) {
        const float xv = xr[k];
#pragma unroll
        for (int e = 0; e < MOE_E; ++e)
            acc[e] += xv * gw[e * MOE_H + k];
    }
#pragma unroll
    for (int off = 16; off > 0; off >>= 1) {
#pragma unroll
        for (int e = 0; e < MOE_E; ++e)
            acc[e] += __shfl_xor(acc[e], off, 32);
    }

    if (lane == 0) {
        int i0 = 0; float m0 = acc[0];
#pragma unroll
        for (int e = 1; e < MOE_E; ++e)
            if (acc[e] > m0) { m0 = acc[e]; i0 = e; }
        int i1 = -1; float m1 = -3.4e38f;
#pragma unroll
        for (int e = 0; e < MOE_E; ++e)
            if (e != i0 && acc[e] > m1) { m1 = acc[e]; i1 = e; }

        const float r0 = __builtin_amdgcn_rcpf(1.0f + __expf(m1 - m0));
        const float r1 = 1.0f - r0;

        rw_out[(size_t)tok * 2 + 0] = r0;
        rw_out[(size_t)tok * 2 + 1] = r1;

        int p0 = atomicAdd(&counts[i0], 1);
        list[(size_t)i0 * T + p0] = tok;  lwgt[(size_t)i0 * T + p0] = r0;
        int p1 = atomicAdd(&counts[i1], 1);
        list[(size_t)i1 * T + p1] = tok;  lwgt[(size_t)i1 * T + p1] = r1;
    }
}

// Build one 16x32 f16 A-fragment from a packed-f16 row (two 16B loads).
__device__ __forceinline__ v16h load_a_frag_f16(const _Float16* __restrict__ row, int kb) {
    union { v16h v; v8h h[2]; } au;
    au.h[0] = *(const v8h*)(row + kb);
    au.h[1] = *(const v8h*)(row + kb + 16);
    return au.v;
}

// ---------------------------------------------------------------------------
// Kernel 2: fused routed SwiGLU FFN for one (expert, 32-token tile).
// 128 threads = 4 wave32. Every B fragment (weights, cvt fp32->f16) feeds
// two 16-row M-tiles -> 2x WMMA per load. h (f16) lives in 256KB of WGP LDS.
// A-fragments come from the pre-converted f16 activation buffer (no cvt).
// ---------------------------------------------------------------------------
__global__ __launch_bounds__(128) void moe_ffn_kernel(
    const _Float16* __restrict__ xh,
    const float* __restrict__ w1, const float* __restrict__ w2,
    const float* __restrict__ w3,
    const int* __restrict__ counts, const int* __restrict__ list,
    const float* __restrict__ lwgt,
    float* __restrict__ out, int T)
{
    const int tilesPerE = T / MOE_M;
    const int e    = blockIdx.x / tilesPerE;
    const int tile = blockIdx.x % tilesPerE;
    const int cnt  = counts[e];
    const int base = tile * MOE_M;
    if (base >= cnt) return;

    __shared__ __align__(32) _Float16 Hs[MOE_M * MOE_F];  // 256 KB activations
    __shared__ int   s_tok[MOE_M];
    __shared__ float s_w[MOE_M];

    const int tid  = threadIdx.x;
    const int lane = tid & 31;
    const int wv   = tid >> 5;

    if (tid < MOE_M) {
        const int idx = base + tid;
        if (idx < cnt) { s_tok[tid] = list[(size_t)e * T + idx];
                         s_w[tid]   = lwgt[(size_t)e * T + idx]; }
        else           { s_tok[tid] = list[(size_t)e * T + base];
                         s_w[tid]   = 0.0f; }
    }
    __syncthreads();

    const float* w1e = w1 + (size_t)e * MOE_H * MOE_F;
    const float* w3e = w3 + (size_t)e * MOE_H * MOE_F;
    const float* w2e = w2 + (size_t)e * MOE_F * MOE_H;

    const int n15   = lane & 15;                 // A: M row / B,C: N col
    const int khalf = (lane >> 4) << 3;          // K sub-offset: 0 or 8
    const int mbase = (lane < 16) ? 0 : 8;       // C/D: M row group

    // per-lane gathered input rows in packed f16 (uniform over the k loop)
    const _Float16* xr_lo = xh + (size_t)s_tok[n15]      * MOE_H;
    const _Float16* xr_hi = xh + (size_t)s_tok[n15 + 16] * MOE_H;

    // ---- Phase 1: h = silu(X @ w1) * (X @ w3); wave wv owns F cols ft..ft+15
    for (int ft = wv * 16; ft < MOE_F; ft += 64) {
        v8f acc1lo = {}, acc1hi = {}, acc3lo = {}, acc3hi = {};
        for (int k0 = 0; k0 < MOE_H; k0 += 32) {
            const int kb = k0 + khalf;
            const v16h alo = load_a_frag_f16(xr_lo, kb);
            const v16h ahi = load_a_frag_f16(xr_hi, kb);

            union { v16h v; _Float16 s[16]; } b1u, b3u;
            const float* p1 = w1e + (size_t)kb * MOE_F + ft + n15;
            const float* p3 = w3e + (size_t)kb * MOE_F + ft + n15;
#pragma unroll
            for (int i = 0; i < 8; ++i) {
                b1u.s[i] = (_Float16)p1[(size_t)i * MOE_F];
                b3u.s[i] = (_Float16)p3[(size_t)i * MOE_F];
            }
#pragma unroll
            for (int i = 0; i < 8; ++i) {
                b1u.s[8 + i] = (_Float16)p1[(size_t)(16 + i) * MOE_F];
                b3u.s[8 + i] = (_Float16)p3[(size_t)(16 + i) * MOE_F];
            }
            // prefetch next k-step's weight rows (global_prefetch_b8)
            __builtin_prefetch(p1 + (size_t)32 * MOE_F, 0, 1);
            __builtin_prefetch(p3 + (size_t)32 * MOE_F, 0, 1);

            acc1lo = __builtin_amdgcn_wmma_f32_16x16x32_f16(
                false, alo, false, b1u.v, (short)0, acc1lo, false, false);
            acc1hi = __builtin_amdgcn_wmma_f32_16x16x32_f16(
                false, ahi, false, b1u.v, (short)0, acc1hi, false, false);
            acc3lo = __builtin_amdgcn_wmma_f32_16x16x32_f16(
                false, alo, false, b3u.v, (short)0, acc3lo, false, false);
            acc3hi = __builtin_amdgcn_wmma_f32_16x16x32_f16(
                false, ahi, false, b3u.v, (short)0, acc3hi, false, false);
        }
#pragma unroll
        for (int v = 0; v < 8; ++v) {
            const float a1l = acc1lo[v], a3l = acc3lo[v];
            const float a1h = acc1hi[v], a3h = acc3hi[v];
            // silu(a)*b with fast reciprocal (no IEEE div chain)
            const float hl = a1l * __builtin_amdgcn_rcpf(1.0f + __expf(-a1l)) * a3l;
            const float hh = a1h * __builtin_amdgcn_rcpf(1.0f + __expf(-a1h)) * a3h;
            const int m = v + mbase;
            Hs[(size_t)m        * MOE_F + ft + n15] = (_Float16)hl;
            Hs[(size_t)(m + 16) * MOE_F + ft + n15] = (_Float16)hh;
        }
    }
    __syncthreads();

    // ---- Phase 2: out_tile = h @ w2; wave wv owns N-chunks {wv, wv+4, ...}
    for (int nc = wv; nc < (MOE_H >> 4); nc += 4) {
        const int nb = nc << 4;
        v8f acc_lo = {}, acc_hi = {};
        for (int k0 = 0; k0 < MOE_F; k0 += 32) {
            const int kb = k0 + khalf;
            const v16h alo = load_a_frag_f16(Hs + (size_t)n15 * MOE_F, kb);
            const v16h ahi = load_a_frag_f16(Hs + (size_t)(n15 + 16) * MOE_F, kb);

            union { v16h v; _Float16 s[16]; } bu;
            const float* p2 = w2e + (size_t)kb * MOE_H + nb + n15;
#pragma unroll
            for (int i = 0; i < 8; ++i) bu.s[i] = (_Float16)p2[(size_t)i * MOE_H];
#pragma unroll
            for (int i = 0; i < 8; ++i) bu.s[8 + i] = (_Float16)p2[(size_t)(16 + i) * MOE_H];
            __builtin_prefetch(p2 + (size_t)32 * MOE_H, 0, 1);

            acc_lo = __builtin_amdgcn_wmma_f32_16x16x32_f16(
                false, alo, false, bu.v, (short)0, acc_lo, false, false);
            acc_hi = __builtin_amdgcn_wmma_f32_16x16x32_f16(
                false, ahi, false, bu.v, (short)0, acc_hi, false, false);
        }
#pragma unroll
        for (int v = 0; v < 8; ++v) {
            const int m = v + mbase;
            const float wl = s_w[m];
            const float wh = s_w[m + 16];
            if (wl != 0.0f)
                atomicAdd(out + (size_t)s_tok[m] * MOE_H + nb + n15, acc_lo[v] * wl);
            if (wh != 0.0f)
                atomicAdd(out + (size_t)s_tok[m + 16] * MOE_H + nb + n15, acc_hi[v] * wh);
        }
    }
}

// ---------------------------------------------------------------------------
// Host launcher
// ---------------------------------------------------------------------------
extern "C" void kernel_launch(void* const* d_in, const int* in_sizes, int n_in,
                              void* d_out, int out_size, void* d_ws, size_t ws_size,
                              hipStream_t stream) {
    const float* x  = (const float*)d_in[0];   // [T, 2048]
    const float* gw = (const float*)d_in[1];   // [8, 2048]
    const float* w1 = (const float*)d_in[2];   // [8, 2048, 4096]
    const float* w2 = (const float*)d_in[3];   // [8, 4096, 2048]
    const float* w3 = (const float*)d_in[4];   // [8, 2048, 4096]

    const int T = in_sizes[0] / MOE_H;         // 4096 tokens
    const size_t nx = (size_t)T * MOE_H;

    float* out = (float*)d_out;                // [T, 2048]
    float* rw  = out + nx;                     // [T, 2] router weights

    // workspace: counts | lists | list-weights | packed f16 activations
    int*      counts = (int*)d_ws;
    int*      list   = (int*)((char*)d_ws + 256);
    float*    lwgt   = (float*)((char*)d_ws + 256 + (size_t)MOE_E * T * sizeof(int));
    _Float16* xh     = (_Float16*)((char*)d_ws + 256 + (size_t)MOE_E * T * 8);

    hipMemsetAsync(d_ws, 0, 256, stream);
    hipMemsetAsync(d_out, 0, nx * sizeof(float), stream);

    cvt_x_kernel<<<(int)(nx / (256 * 8)), 256, 0, stream>>>(x, xh, nx);
    moe_gate_kernel<<<T / 8, 256, 0, stream>>>(x, gw, rw, counts, list, lwgt, T);

    const int blocks = MOE_E * (T / MOE_M);    // 1024 blocks; inactive tiles exit early
    moe_ffn_kernel<<<blocks, 128, 0, stream>>>(xh, w1, w2, w3, counts, list, lwgt, out, T);
}
